// ReMoGNN_74577812128298
// MI455X (gfx1250) — matrix operand order repaired and checked by reading, not compiled, not run on hardware
//
#include <hip/hip_runtime.h>

typedef __attribute__((ext_vector_type(16))) __bf16 v16bf;
typedef __attribute__((ext_vector_type(8)))  float  v8f;

__device__ __forceinline__ unsigned short f2bf(float f) {
    unsigned int u = __float_as_uint(f);
    u += 0x7FFFu + ((u >> 16) & 1u);          // round-to-nearest-even
    return (unsigned short)(u >> 16);
}

union AFrag { v16bf v; unsigned int u[8]; };

// ---------------------------------------------------------------------------
// bf16 WMMA GEMM: C[M,Nout] = A[M,K] * B[K,Nout] (+bias, optional relu)
// Nout is a multiple of 128 (covered by gridDim.y), K a multiple of 32.
// Block = 256 threads = 8 waves; block tile 128x128; wave tile 32x64.
// ---------------------------------------------------------------------------
__global__ __launch_bounds__(256)
void gemm_bf16_wmma(const float* __restrict__ A, int lda,
                    const float* __restrict__ B, int ldb,
                    float* __restrict__ C, int ldc,
                    int M, int K,
                    const float* __restrict__ bias, int relu)
{
    __shared__ unsigned short As[128][34];   // [m][k] bf16 bits, +2 pad
    __shared__ unsigned short Bs[128][34];   // [n][k] bf16 bits (transposed), +2 pad

    const int tid  = threadIdx.x;
    const int m0   = blockIdx.x * 128;
    const int n0   = blockIdx.y * 128;
    const int lane = tid & 31;
    const int wave = tid >> 5;
    const int wy   = wave >> 1;              // 0..3 (row of wave grid)
    const int wx   = wave & 1;               // 0..1 (col of wave grid)
    const int lm   = lane & 15;
    const int hi   = lane >> 4;              // 0 or 1

    v8f acc[2][4];
    for (int s = 0; s < 2; ++s)
        for (int t = 0; t < 4; ++t)
            for (int i = 0; i < 8; ++i) acc[s][t][i] = 0.0f;

    for (int k0 = 0; k0 < K; k0 += 32) {
        // --- stage A tile (128x32 fp32 -> bf16) ---
        {
            const int row = tid >> 3;            // 0..31
            const int cg  = (tid & 7) * 4;       // 0..28
            #pragma unroll
            for (int p = 0; p < 4; ++p) {
                const int r  = p * 32 + row;
                const int gr = m0 + r;
                float4 f = make_float4(0.f, 0.f, 0.f, 0.f);
                if (gr < M)
                    f = *reinterpret_cast<const float4*>(A + (long long)gr * lda + k0 + cg);
                As[r][cg + 0] = f2bf(f.x);
                As[r][cg + 1] = f2bf(f.y);
                As[r][cg + 2] = f2bf(f.z);
                As[r][cg + 3] = f2bf(f.w);
            }
        }
        // --- stage B tile (32x128 fp32 -> bf16, transposed to [n][k]) ---
        {
            const int krow = tid >> 5;           // 0..7
            const int ng   = (tid & 31) * 4;     // 0..124
            #pragma unroll
            for (int p = 0; p < 4; ++p) {
                const int kr = p * 8 + krow;
                float4 f = *reinterpret_cast<const float4*>(
                    B + (long long)(k0 + kr) * ldb + n0 + ng);
                Bs[ng + 0][kr] = f2bf(f.x);
                Bs[ng + 1][kr] = f2bf(f.y);
                Bs[ng + 2][kr] = f2bf(f.z);
                Bs[ng + 3][kr] = f2bf(f.w);
            }
        }
        __syncthreads();

        // --- fragments per ISA 7.12.2 layouts ---
        AFrag af[2], bfr[4];
        #pragma unroll
        for (int s = 0; s < 2; ++s) {
            const unsigned int* ar =
                reinterpret_cast<const unsigned int*>(&As[wy * 32 + s * 16 + lm][0]);
            const int kb2 = hi * 4;              // K pair base (lanes>=16 -> K+8)
            #pragma unroll
            for (int j = 0; j < 4; ++j) {
                af[s].u[j]     = ar[kb2 + j];        // K = hi*8 + 2j   (.. +1)
                af[s].u[4 + j] = ar[8 + kb2 + j];    // K = 16 + hi*8 + 2j
            }
        }
        #pragma unroll
        for (int t = 0; t < 4; ++t) {
            const unsigned int* br =
                reinterpret_cast<const unsigned int*>(&Bs[wx * 64 + t * 16 + lm][0]);
            #pragma unroll
            for (int j = 0; j < 8; ++j)
                bfr[t].u[j] = br[(j < 4) ? (hi * 4 + j) : (8 + hi * 4 + (j - 4))];        // K = hi*8 + 2j (j<4), 16 + hi*8 + 2(j-4) (j>=4) (.. +1)
        }

        #pragma unroll
        for (int s = 0; s < 2; ++s)
            #pragma unroll
            for (int t = 0; t < 4; ++t)
                acc[s][t] = __builtin_amdgcn_wmma_f32_16x16x32_bf16(
                    false, af[s].v, false, bfr[t].v, (short)0, acc[s][t], false, false);

        __syncthreads();
    }

    // --- epilogue: D 16x16 f32 layout (lane col = lane&15, rows hi*8+v) ---
    #pragma unroll
    for (int s = 0; s < 2; ++s) {
        const int rbase = m0 + wy * 32 + s * 16 + hi * 8;
        #pragma unroll
        for (int t = 0; t < 4; ++t) {
            const int col = n0 + wx * 64 + t * 16 + lm;
            const float bv = bias ? bias[col] : 0.0f;
            #pragma unroll
            for (int v = 0; v < 8; ++v) {
                const int rowg = rbase + v;
                if (rowg < M) {
                    float val = acc[s][t][v] + bv;
                    if (relu) val = fmaxf(val, 0.0f);
                    C[(long long)rowg * ldc + col] = val;
                }
            }
        }
    }
}

// ---------------------------------------------------------------------------
// Helper kernels
// ---------------------------------------------------------------------------
__device__ __forceinline__ void atomAddF(float* p, float v) {
    __hip_atomic_fetch_add(p, v, __ATOMIC_RELAXED, __HIP_MEMORY_SCOPE_AGENT);
}

__global__ void zero_kernel(float* __restrict__ p, long long n) {
    long long g = (long long)blockIdx.x * blockDim.x + threadIdx.x;
    if (g < n) p[g] = 0.0f;
}

// pack [c1_wrel (3*128 rows) ; c1_wroot (128 rows)] -> Wpack[512,768]
__global__ void pack_w1_kernel(const float* __restrict__ wrel,
                               const float* __restrict__ wroot,
                               float* __restrict__ wpk) {
    long long g = (long long)blockIdx.x * blockDim.x + threadIdx.x;
    const long long nrel = (long long)384 * 768;
    if (g < nrel) wpk[g] = wrel[g];
    else if (g < nrel + (long long)128 * 768) wpk[g] = wroot[g - nrel];
}

// per-(node,relation) in-degree counts (float)
__global__ void count_kernel(const int* __restrict__ tgt,
                             const int* __restrict__ et,
                             float* __restrict__ cnt, int E) {
    int e = blockIdx.x * blockDim.x + threadIdx.x;
    if (e < E) atomAddF(&cnt[(long long)tgt[e] * 3 + et[e]], 1.0f);
}

// scatter-add 128-wide features into per-(node,rel) slots.
// feat row stride = fstride; source col offset = rel*srcRelStride.
// out[tgt*ostride + rel*128 + j] += feat[src*fstride + rel*srcRelStride + j]
__global__ void scatter_kernel(const int* __restrict__ src,
                               const int* __restrict__ tgt,
                               const int* __restrict__ et,
                               const float* __restrict__ feat, int fstride, int srcRelStride,
                               float* __restrict__ out, int ostride, int E) {
    long long g = (long long)blockIdx.x * blockDim.x + threadIdx.x;
    if (g >= (long long)E * 128) return;
    const int e = (int)(g >> 7);
    const int j = (int)(g & 127);
    const int r = et[e];
    const float v = feat[(long long)src[e] * fstride + r * srcRelStride + j];
    atomAddF(&out[(long long)tgt[e] * ostride + (long long)r * 128 + j], v);
}

// copy q into columns [384,512) of the layer-1 concat matrix (stride 512)
__global__ void copyq_kernel(const float* __restrict__ q, float* __restrict__ acat,
                             long long N) {
    long long g = (long long)blockIdx.x * blockDim.x + threadIdx.x;
    if (g >= N * 128) return;
    const long long n = g >> 7;
    const int j = (int)(g & 127);
    acat[n * 512 + 384 + j] = q[g];
}

// divide columns [0,384) of acat (stride 512) by max(cnt,1)
__global__ void normalize_kernel(float* __restrict__ acat,
                                 const float* __restrict__ cnt, long long N) {
    long long g = (long long)blockIdx.x * blockDim.x + threadIdx.x;
    if (g >= N * 384) return;
    const long long n = g / 384;
    const int c = (int)(g % 384);
    const int r = c >> 7;
    const float d = cnt[n * 3 + r];
    acat[n * 512 + c] *= 1.0f / fmaxf(d, 1.0f);
}

// h2[n,j] = relu(h2root[n,j] + sum_r agg2[n,r,j]/max(cnt,1))   (in place on h2)
__global__ void combine2_kernel(float* __restrict__ h2,
                                const float* __restrict__ agg2,
                                const float* __restrict__ cnt, long long N) {
    long long g = (long long)blockIdx.x * blockDim.x + threadIdx.x;
    if (g >= N * 128) return;
    const long long n = g >> 7;
    const int j = (int)(g & 127);
    float v = h2[g];
    #pragma unroll
    for (int r = 0; r < 3; ++r) {
        const float d = cnt[n * 3 + r];
        v += agg2[n * 384 + (long long)r * 128 + j] * (1.0f / fmaxf(d, 1.0f));
    }
    h2[g] = fmaxf(v, 0.0f);
}

// ---------------------------------------------------------------------------
// Orchestration
// ---------------------------------------------------------------------------
extern "C" void kernel_launch(void* const* d_in, const int* in_sizes, int n_in,
                              void* d_out, int out_size, void* d_ws, size_t ws_size,
                              hipStream_t stream) {
    const float* x       = (const float*)d_in[0];
    const int*   eidx    = (const int*)d_in[1];
    const int*   etype   = (const int*)d_in[2];
    const float* q_w1    = (const float*)d_in[3];
    const float* q_b1    = (const float*)d_in[4];
    const float* q_w2    = (const float*)d_in[5];
    const float* q_b2    = (const float*)d_in[6];
    const float* c1_wrel = (const float*)d_in[7];
    const float* c1_wroot= (const float*)d_in[8];
    const float* c1_b    = (const float*)d_in[9];
    const float* c2_wrel = (const float*)d_in[10];
    const float* c2_wroot= (const float*)d_in[11];
    const float* c2_b    = (const float*)d_in[12];
    const float* lin_w   = (const float*)d_in[13];
    const float* lin_b   = (const float*)d_in[14];
    float* out = (float*)d_out;

    const long long N = in_sizes[0] / 128;
    const int E = in_sizes[2];
    const int* src = eidx;
    const int* tgt = eidx + E;

    // workspace layout (floats)
    float* ws   = (float*)d_ws;
    float* hid  = ws;                    // [N,768] MLP hidden; reused as h1
    float* q    = hid  + N * 768;        // [N,128] q; reused as h2root/h2
    float* acat = q    + N * 128;        // [N,512] layer-1 concat; reused as t [N,384]
    float* agg2 = acat + N * 512;        // [N,384]
    float* cnt  = agg2 + N * 384;        // [N,3]
    float* wpk  = cnt  + N * 3;          // [512,768]

    const dim3 blk(256);
    const int  MB = (int)((N + 127) / 128);   // 391 row tiles

    // 0) zero the accumulation span (acat + agg2 + cnt are contiguous)
    {
        const long long nz = N * (512 + 384 + 3);
        zero_kernel<<<dim3((unsigned)((nz + 255) / 256)), blk, 0, stream>>>(acat, nz);
    }
    // pack layer-1 weights [c1_wrel ; c1_wroot] -> [512,768]
    {
        const long long np = (long long)512 * 768;
        pack_w1_kernel<<<dim3((unsigned)((np + 255) / 256)), blk, 0, stream>>>(c1_wrel, c1_wroot, wpk);
    }
    // per-(node,rel) edge counts (shared by both layers)
    count_kernel<<<dim3((E + 255) / 256), blk, 0, stream>>>(tgt, etype, cnt, E);

    // 1) query MLP
    gemm_bf16_wmma<<<dim3(MB, 6), blk, 0, stream>>>(x, 128, q_w1, 768, hid, 768,
                                                    (int)N, 128, q_b1, 1);
    gemm_bf16_wmma<<<dim3(MB, 1), blk, 0, stream>>>(hid, 768, q_w2, 128, q, 128,
                                                    (int)N, 768, q_b2, 0);

    // 2) layer 1: aggregate-then-transform (means commute with W_r)
    {
        const long long ng = (long long)E * 128;
        scatter_kernel<<<dim3((unsigned)((ng + 255) / 256)), blk, 0, stream>>>(
            src, tgt, etype, q, 128, 0, acat, 512, E);
    }
    {
        const long long nc = N * 128;
        copyq_kernel<<<dim3((unsigned)((nc + 255) / 256)), blk, 0, stream>>>(q, acat, N);
    }
    {
        const long long nn = N * 384;
        normalize_kernel<<<dim3((unsigned)((nn + 255) / 256)), blk, 0, stream>>>(acat, cnt, N);
    }
    // h1 = relu(acat[N,512] @ Wpack[512,768] + c1_b) -> hid
    gemm_bf16_wmma<<<dim3(MB, 6), blk, 0, stream>>>(acat, 512, wpk, 768, hid, 768,
                                                    (int)N, 512, c1_b, 1);

    // 3) layer 2: transform-then-gather (output side is only 128 wide)
    for (int r = 0; r < 3; ++r) {
        gemm_bf16_wmma<<<dim3(MB, 1), blk, 0, stream>>>(
            hid, 768, c2_wrel + (long long)r * 768 * 128, 128,
            acat + r * 128, 384, (int)N, 768, nullptr, 0);   // t[:,r,:]
    }
    {
        const long long ng = (long long)E * 128;
        scatter_kernel<<<dim3((unsigned)((ng + 255) / 256)), blk, 0, stream>>>(
            src, tgt, etype, acat, 384, 128, agg2, 384, E);
    }
    // root transform -> q buffer (h2root), bias included
    gemm_bf16_wmma<<<dim3(MB, 1), blk, 0, stream>>>(hid, 768, c2_wroot, 128, q, 128,
                                                    (int)N, 768, c2_b, 0);
    {
        const long long nc = N * 128;
        combine2_kernel<<<dim3((unsigned)((nc + 255) / 256)), blk, 0, stream>>>(q, agg2, cnt, N);
    }

    // 4) final linear -> d_out
    gemm_bf16_wmma<<<dim3(MB, 1), blk, 0, stream>>>(q, 128, lin_w, 128, out, 128,
                                                    (int)N, 128, lin_b, 0);
}
